// TimeMamba_52003464020735
// MI455X (gfx1250) — compile-verified, matrix operand order checked by reading, not run on hardware
//
#include <hip/hip_runtime.h>
#include <hip/hip_bf16.h>
#include <math.h>

// ---------------- problem constants ----------------
#define D_MODEL 512
#define D_STATE 16
#define D_CONV  4
#define D_INNER 1024
#define DT_RANK 32
#define BZ      2048
#define SEQ     5
#define ROWS    (BZ * SEQ)        // 10240 GEMM rows
#define XPROJ_N 128               // 64 real + 64 zero-pad (keeps N % 128 == 0)

// ---------------- vector types ----------------
typedef __attribute__((ext_vector_type(16))) __bf16 bf16x16;
typedef __attribute__((ext_vector_type(8)))  __bf16 bf16x8;
typedef __attribute__((ext_vector_type(8)))  float  f32x8;
typedef __attribute__((ext_vector_type(4)))  float  f32x4;

// ---------------- fast activation helpers ----------------
__device__ __forceinline__ float fast_sigmoid(float x) {
    return __builtin_amdgcn_rcpf(1.f + __expf(-x));        // v_rcp + v_exp
}
__device__ __forceinline__ float fast_softplus(float x) {
    // stable: max(x,0) + log(1 + exp(-|x|)), hardware exp/log
    return fmaxf(x, 0.f) + __logf(1.f + __expf(-fabsf(x)));
}

// =====================================================================
// WMMA GEMM:  C[M,N] = A[M,K](f32,row-major) * B[N,K](bf16,row-major)^T
// block tile 256x128, 8 waves (4 x 2), wave tile 64x64 (16 WMMA accums),
// BK = 32, double-buffered LDS, one barrier per K-step.
// mode 0: plain store; mode 1: softplus(acc + bias[col]) (dt projection).
// M % 256 == 0 (10240), N % 128 == 0, K % 32 == 0.
// =====================================================================
#define BM 256
#define BN 128
#define BK 32
#define SKEW 8
#define LDSTR (BK + SKEW)   // 40 bf16 = 80B row stride (16B multiple)

__device__ __forceinline__ void stage_tiles(__bf16* __restrict__ sAb,
                                            __bf16* __restrict__ sBb,
                                            int tid,
                                            const f32x4* areg,
                                            const bf16x8* breg) {
    __bf16* sp = sAb + tid * LDSTR;             // thread = one A row (32 cols)
    #pragma unroll
    for (int j = 0; j < 4; ++j) {
        bf16x8 h;
        h[0] = (__bf16)areg[2*j][0];   h[1] = (__bf16)areg[2*j][1];
        h[2] = (__bf16)areg[2*j][2];   h[3] = (__bf16)areg[2*j][3];
        h[4] = (__bf16)areg[2*j+1][0]; h[5] = (__bf16)areg[2*j+1][1];
        h[6] = (__bf16)areg[2*j+1][2]; h[7] = (__bf16)areg[2*j+1][3];
        *(bf16x8*)(sp + 8 * j) = h;
    }
    __bf16* spb = sBb + (tid >> 1) * LDSTR + (tid & 1) * 16;
    *(bf16x8*)(spb)     = breg[0];
    *(bf16x8*)(spb + 8) = breg[1];
}

__global__ __launch_bounds__(256) void wmma_gemm_kernel(
    const float* __restrict__ A, int lda,
    const __bf16* __restrict__ B,            // ldb == K
    float* __restrict__ C, int ldc,
    int K, const float* __restrict__ bias, int mode)
{
    __shared__ __bf16 sA[2][BM * LDSTR];     // 40 KB
    __shared__ __bf16 sB[2][BN * LDSTR];     // 20 KB

    const int tid  = threadIdx.x;
    const int lane = tid & 31;
    const int wave = tid >> 5;       // 0..7
    const int wm   = wave >> 1;      // 0..3  (M direction, 64 rows each)
    const int wn   = wave & 1;       // 0..1  (N direction, 64 cols each)

    const int blockM = blockIdx.y * BM;
    const int blockN = blockIdx.x * BN;

    // fragment addressing: lane&15 = row/col in 16-tile, lane>>4 = K-half
    const int frow  = lane & 15;
    const int khalf = lane >> 4;

    const float*  aptr = A + (size_t)(blockM + tid) * lda;              // 1 row/thread
    const __bf16* bptr = B + (size_t)(blockN + (tid >> 1)) * K + (tid & 1) * 16;

    f32x8 acc[4][4] = {};
    f32x4 areg[8];
    bf16x8 breg[2];

    // ---- prologue: fetch + stage tile 0 ----
    #pragma unroll
    for (int j = 0; j < 8; ++j) areg[j] = *(const f32x4*)(aptr + 4 * j);
    breg[0] = *(const bf16x8*)(bptr);
    breg[1] = *(const bf16x8*)(bptr + 8);
    stage_tiles(sA[0], sB[0], tid, areg, breg);
    __syncthreads();

    const int nT = K / BK;
    for (int t = 0; t < nT; ++t) {
        const int cur = t & 1;
        const bool more = (t + 1 < nT);
        // prefetch next tile into registers while WMMAs consume LDS[cur]
        if (more) {
            const float* ap = aptr + (t + 1) * BK;
            #pragma unroll
            for (int j = 0; j < 8; ++j) areg[j] = *(const f32x4*)(ap + 4 * j);
            const __bf16* bp = bptr + (t + 1) * BK;
            breg[0] = *(const bf16x8*)(bp);
            breg[1] = *(const bf16x8*)(bp + 8);
        }
        // fragment loads (CDNA5 16-bit layout: K 0..7|16..23 lanes 0-15,
        // K 8..15|24..31 lanes 16-31)
        bf16x16 af[4], bfm[4];
        #pragma unroll
        for (int mi = 0; mi < 4; ++mi) {
            const __bf16* p = sA[cur] + (wm * 64 + mi * 16 + frow) * LDSTR + khalf * 8;
            *( (bf16x8*)&af[mi] )     = *(const bf16x8*)(p);
            *(((bf16x8*)&af[mi]) + 1) = *(const bf16x8*)(p + 16);
        }
        #pragma unroll
        for (int ni = 0; ni < 4; ++ni) {
            const __bf16* p = sB[cur] + (wn * 64 + ni * 16 + frow) * LDSTR + khalf * 8;
            *( (bf16x8*)&bfm[ni] )     = *(const bf16x8*)(p);
            *(((bf16x8*)&bfm[ni]) + 1) = *(const bf16x8*)(p + 16);
        }
        #pragma unroll
        for (int mi = 0; mi < 4; ++mi)
            #pragma unroll
            for (int ni = 0; ni < 4; ++ni)
                acc[mi][ni] = __builtin_amdgcn_wmma_f32_16x16x32_bf16(
                    false, af[mi], false, bfm[ni], (short)0, acc[mi][ni],
                    false, false);
        // stage next tile into the other buffer (no conflict with readers of cur)
        if (more) stage_tiles(sA[cur ^ 1], sB[cur ^ 1], tid, areg, breg);
        __syncthreads();
    }

    // ---- store: C/D layout VGPR i -> row i (lanes 0-15) / row i+8 (16-31) ----
    const int rbase = blockM + wm * 64 + (khalf << 3);
    const int cbase = blockN + wn * 64 + frow;
    #pragma unroll
    for (int mi = 0; mi < 4; ++mi) {
        #pragma unroll
        for (int ni = 0; ni < 4; ++ni) {
            const int col = cbase + ni * 16;
            #pragma unroll
            for (int i = 0; i < 8; ++i) {
                float v = acc[mi][ni][i];
                if (mode == 1) v = fast_softplus(v + bias[col]);
                C[(size_t)(rbase + mi * 16 + i) * ldc + col] = v;
            }
        }
    }
}

// =====================================================================
// Weight prep kernels
// =====================================================================
__global__ void cvt_bf16_kernel(const float* __restrict__ src,
                                __bf16* __restrict__ dst, int n) {
    int i = blockIdx.x * blockDim.x + threadIdx.x;
    if (i < n) dst[i] = (__bf16)src[i];
}

__global__ void pad_xproj_kernel(const float* __restrict__ src,
                                 __bf16* __restrict__ dst) {
    int i = blockIdx.x * blockDim.x + threadIdx.x;   // 128*1024
    if (i >= XPROJ_N * D_INNER) return;
    int r = i >> 10, c = i & (D_INNER - 1);
    dst[i] = (r < (DT_RANK + 2 * D_STATE)) ? (__bf16)src[r * D_INNER + c]
                                           : (__bf16)0.f;
}

__global__ void aneg_kernel(const float* __restrict__ alog,
                            float* __restrict__ aneg, int n) {
    int i = blockIdx.x * blockDim.x + threadIdx.x;
    if (i < n) aneg[i] = -__expf(alog[i]);
}

// =====================================================================
// Depthwise causal conv (K=4) + SiLU over seq (L=5); thread per (b,d)
// =====================================================================
__global__ void conv_silu_kernel(const float* __restrict__ xz,
                                 const float* __restrict__ wconv,
                                 const float* __restrict__ bconv,
                                 float* __restrict__ xc) {
    int idx = blockIdx.x * blockDim.x + threadIdx.x;
    if (idx >= BZ * D_INNER) return;
    int b = idx >> 10, d = idx & (D_INNER - 1);
    float w[D_CONV];
    #pragma unroll
    for (int j = 0; j < D_CONV; ++j) w[j] = wconv[d * D_CONV + j];
    const float bb = bconv[d];
    const float* xi = xz + (size_t)b * SEQ * (2 * D_INNER) + d;
    float x[SEQ];
    #pragma unroll
    for (int l = 0; l < SEQ; ++l) x[l] = xi[l * (2 * D_INNER)];
    #pragma unroll
    for (int l = 0; l < SEQ; ++l) {
        float a = bb;
        #pragma unroll
        for (int j = 0; j < D_CONV; ++j) {
            int s = l - (D_CONV - 1) + j;
            if (s >= 0) a += w[j] * x[s];
        }
        xc[((size_t)b * SEQ + l) * D_INNER + d] = a * fast_sigmoid(a);   // silu
    }
}

// =====================================================================
// Selective scan; thread per (b,d); h[16] in VGPRs
// =====================================================================
__global__ void scan_kernel(const float* __restrict__ xz,
                            const float* __restrict__ xc,
                            const float* __restrict__ dbl,
                            const float* __restrict__ dtb,
                            const float* __restrict__ aneg,
                            const float* __restrict__ Dp,
                            float* __restrict__ yb) {
    int idx = blockIdx.x * blockDim.x + threadIdx.x;
    if (idx >= BZ * D_INNER) return;
    int b = idx >> 10, d = idx & (D_INNER - 1);
    float An[D_STATE];
    #pragma unroll
    for (int n = 0; n < D_STATE; ++n) An[n] = aneg[d * D_STATE + n];
    const float Dd = Dp[d];
    float h[D_STATE];
    #pragma unroll
    for (int n = 0; n < D_STATE; ++n) h[n] = 0.f;

    for (int l = 0; l < SEQ; ++l) {
        const size_t bl = (size_t)b * SEQ + l;
        const float dt  = dtb[bl * D_INNER + d];
        const float xcv = xc[bl * D_INNER + d];
        const float* bc = dbl + bl * XPROJ_N;   // [32..47]=B, [48..63]=C
        float y = 0.f;
        #pragma unroll
        for (int n = 0; n < D_STATE; ++n) {
            float dA = __expf(dt * An[n]);
            h[n] = dA * h[n] + dt * bc[DT_RANK + n] * xcv;
            y += h[n] * bc[DT_RANK + D_STATE + n];
        }
        y += Dd * xcv;
        float z = xz[bl * (2 * D_INNER) + D_INNER + d];
        y *= z * fast_sigmoid(z);                             // * silu(z)
        yb[bl * D_INNER + d] = y;
    }
}

// =====================================================================
// Fusion: atten = max_c + mean_c ; wgt = softmax(atten @ W^T + b);
// out[b,:] (+)= sum_s feat[b,s,:] * wgt[s].  One block per batch.
// =====================================================================
template <int S>
__global__ __launch_bounds__(256) void fuse_kernel(
    const float* __restrict__ feat, const float* __restrict__ W,
    const float* __restrict__ bias, float* __restrict__ out, int add) {
    const int b = blockIdx.x;
    const int tid = threadIdx.x;
    __shared__ float smax[256];
    __shared__ float ssum[256];
    __shared__ float att[S];
    __shared__ float wgt[S];
    const float* fb = feat + (size_t)b * S * D_MODEL;

    for (int l = 0; l < S; ++l) {
        float mx = -3.4e38f, sm = 0.f;
        for (int c = tid; c < D_MODEL; c += 256) {
            float v = fb[l * D_MODEL + c];
            mx = fmaxf(mx, v);
            sm += v;
        }
        smax[tid] = mx; ssum[tid] = sm;
        __syncthreads();
        for (int s = 128; s > 0; s >>= 1) {
            if (tid < s) {
                smax[tid] = fmaxf(smax[tid], smax[tid + s]);
                ssum[tid] += ssum[tid + s];
            }
            __syncthreads();
        }
        if (tid == 0) att[l] = smax[0] + ssum[0] * (1.f / (float)D_MODEL);
        __syncthreads();
    }
    if (tid == 0) {
        float lg[S]; float mx = -3.4e38f;
        for (int s = 0; s < S; ++s) {
            float a = bias[s];
            for (int l = 0; l < S; ++l) a += att[l] * W[s * S + l];
            lg[s] = a; mx = fmaxf(mx, a);
        }
        float sum = 0.f;
        for (int s = 0; s < S; ++s) { lg[s] = __expf(lg[s] - mx); sum += lg[s]; }
        float r = __builtin_amdgcn_rcpf(sum);
        for (int s = 0; s < S; ++s) wgt[s] = lg[s] * r;
    }
    __syncthreads();
    for (int c = tid; c < D_MODEL; c += 256) {
        float v = 0.f;
        #pragma unroll
        for (int s = 0; s < S; ++s) v += fb[s * D_MODEL + c] * wgt[s];
        float* o = out + (size_t)b * D_MODEL + c;
        if (add) *o += v; else *o = v;
    }
}

// =====================================================================
// host-side orchestration
// =====================================================================
extern "C" void kernel_launch(void* const* d_in, const int* in_sizes, int n_in,
                              void* d_out, int out_size, void* d_ws, size_t ws_size,
                              hipStream_t stream) {
    (void)in_sizes; (void)n_in; (void)out_size; (void)ws_size;
    const float* img    = (const float*)d_in[0];
    const float* lid    = (const float*)d_in[1];
    const float* rad    = (const float*)d_in[2];
    const float* gps    = (const float*)d_in[3];
    const float* w_in   = (const float*)d_in[4];
    const float* w_conv = (const float*)d_in[5];
    const float* b_conv = (const float*)d_in[6];
    const float* w_xp   = (const float*)d_in[7];
    const float* w_dt   = (const float*)d_in[8];
    const float* b_dt   = (const float*)d_in[9];
    const float* A_log  = (const float*)d_in[10];
    const float* D_par  = (const float*)d_in[11];
    const float* w_out  = (const float*)d_in[12];
    const float* w_mlp  = (const float*)d_in[13];
    const float* b_mlp  = (const float*)d_in[14];
    const float* w_gps  = (const float*)d_in[15];
    const float* b_gps  = (const float*)d_in[16];
    float* out = (float*)d_out;

    // workspace carve (reused across the 3 modalities; stream-serialized)
    char* ws = (char*)d_ws;
    size_t off = 0;
    auto carve = [&](size_t bytes) -> void* {
        void* p = ws + off;
        off = (off + bytes + 255) & ~(size_t)255;
        return p;
    };
    __bf16* wbf_in  = (__bf16*)carve((size_t)(2 * D_INNER) * D_MODEL * 2);
    __bf16* wbf_xp  = (__bf16*)carve((size_t)XPROJ_N * D_INNER * 2);
    __bf16* wbf_dt  = (__bf16*)carve((size_t)D_INNER * DT_RANK * 2);
    __bf16* wbf_out = (__bf16*)carve((size_t)D_MODEL * D_INNER * 2);
    float*  aneg    = (float*)carve((size_t)D_INNER * D_STATE * 4);
    float*  xz      = (float*)carve((size_t)ROWS * (2 * D_INNER) * 4);
    float*  xc      = (float*)carve((size_t)ROWS * D_INNER * 4);
    float*  dbl     = (float*)carve((size_t)ROWS * XPROJ_N * 4);
    float*  dtb     = (float*)carve((size_t)ROWS * D_INNER * 4);
    float*  yb      = (float*)carve((size_t)ROWS * D_INNER * 4);
    float*  mout    = (float*)carve((size_t)ROWS * D_MODEL * 4);

    const int T = 256;
    // ---- weight prep ----
    cvt_bf16_kernel<<<(2 * D_INNER * D_MODEL + T - 1) / T, T, 0, stream>>>(
        w_in, wbf_in, 2 * D_INNER * D_MODEL);
    pad_xproj_kernel<<<(XPROJ_N * D_INNER + T - 1) / T, T, 0, stream>>>(w_xp, wbf_xp);
    cvt_bf16_kernel<<<(D_INNER * DT_RANK + T - 1) / T, T, 0, stream>>>(
        w_dt, wbf_dt, D_INNER * DT_RANK);
    cvt_bf16_kernel<<<(D_MODEL * D_INNER + T - 1) / T, T, 0, stream>>>(
        w_out, wbf_out, D_MODEL * D_INNER);
    aneg_kernel<<<(D_INNER * D_STATE + T - 1) / T, T, 0, stream>>>(
        A_log, aneg, D_INNER * D_STATE);

    // ---- GPS fusion initializes d_out ----
    fuse_kernel<2><<<BZ, T, 0, stream>>>(gps, w_gps, b_gps, out, 0);

    const float* xs[3] = {img, lid, rad};
    const int mtiles = ROWS / BM;   // 40
    for (int m = 0; m < 3; ++m) {
        // 1) xz = x @ w_in^T           (10240 x 512) * (2048 x 512)^T
        wmma_gemm_kernel<<<dim3(2 * D_INNER / BN, mtiles), T, 0, stream>>>(
            xs[m], D_MODEL, wbf_in, xz, 2 * D_INNER, D_MODEL, nullptr, 0);
        // 2) depthwise conv + silu
        conv_silu_kernel<<<(BZ * D_INNER + T - 1) / T, T, 0, stream>>>(
            xz, w_conv, b_conv, xc);
        // 3) dbl = xc @ w_xproj^T      (padded N=128)
        wmma_gemm_kernel<<<dim3(XPROJ_N / BN, mtiles), T, 0, stream>>>(
            xc, D_INNER, wbf_xp, dbl, XPROJ_N, D_INNER, nullptr, 0);
        // 4) dt = softplus(dt_low @ w_dt^T + b_dt)   (K = 32, A = dbl[:, :32])
        wmma_gemm_kernel<<<dim3(D_INNER / BN, mtiles), T, 0, stream>>>(
            dbl, XPROJ_N, wbf_dt, dtb, D_INNER, DT_RANK, b_dt, 1);
        // 5) selective scan + D-skip + silu(z) gate
        scan_kernel<<<(BZ * D_INNER + T - 1) / T, T, 0, stream>>>(
            xz, xc, dbl, dtb, aneg, D_par, yb);
        // 6) mout = y @ w_out^T
        wmma_gemm_kernel<<<dim3(D_MODEL / BN, mtiles), T, 0, stream>>>(
            yb, D_INNER, wbf_out, mout, D_MODEL, D_INNER, nullptr, 0);
        // 7) fuse into d_out
        fuse_kernel<5><<<BZ, T, 0, stream>>>(mout, w_mlp, b_mlp, out, 1);
    }
}
